// EncoderLayer_24601572672000
// MI455X (gfx1250) — matrix operand order repaired and checked by reading, not compiled, not run on hardware
//
#include <hip/hip_runtime.h>

typedef __bf16 bf16_t;
typedef bf16_t v16bf __attribute__((ext_vector_type(16)));
typedef bf16_t v8bf  __attribute__((ext_vector_type(8)));
typedef float  v8f   __attribute__((ext_vector_type(8)));

// ---------------- problem constants (fixed by the harness) ----------------
constexpr int Bn   = 2;
constexpr int LQ   = 10000;        // 100 x 100 BEV
constexpr int Dm   = 256;
constexpr int DFF  = 1024;
constexpr int NH   = 8;
constexpr int HD   = 32;           // Dm / NH
constexpr int BEVH = 100;
constexpr int BEVW = 100;
constexpr int LV   = 19560;        // 92*160+46*80+23*40+12*20
constexpr int Mq   = Bn * LQ;      // 20000 rows
constexpr int Mv   = Bn * LV;      // 39120 rows

// ======================= generic bf16 WMMA GEMM ==========================
// C[M,N] = A[M,K] @ W[K,N] (+bias) (+resid) (relu?) ; out f32 and/or bf16.
// One wave -> 16 x (16*NT) tile; software-pipelined one k-step ahead.
template<int NT>
__global__ __launch_bounds__(256)
void gemm_bf16_kernel(const bf16_t* __restrict__ A,
                      const bf16_t* __restrict__ Wt,
                      const float*  __restrict__ bias,
                      const float*  __restrict__ resid,
                      float*        __restrict__ Cf,
                      bf16_t*       __restrict__ Cb,
                      int Mr, int K, int N, int relu)
{
    const int wave = threadIdx.x >> 5;
    const int lane = threadIdx.x & 31;
    const int m0   = (blockIdx.x * 8 + wave) * 16;
    if (m0 >= Mr) return;                       // wave-uniform -> EXEC stays all-ones
    const int n0   = blockIdx.y * (16 * NT);
    const int half = lane >> 4;
    const int lm   = lane & 15;

    v8f acc[NT];
#pragma unroll
    for (int t = 0; t < NT; ++t)
#pragma unroll
        for (int i = 0; i < 8; ++i) acc[t][i] = 0.0f;

    union V16 { v16bf v; v8bf h[2]; };

    // A fragment (16-bit A 16x32 layout): lane<16 K in [0,8)+[16,24),
    // lane>=16 K in [8,16)+[24,32) -> two contiguous 16B loads per step.
    const bf16_t* Ap = A + (size_t)(m0 + lm) * K + half * 8;
    // B fragment: lane holds column (lane&15), K in same order as A (strided loads).
    const bf16_t* Wp = Wt + (size_t)(half * 8) * N + n0 + lm;
    const size_t  wstep = (size_t)32 * N;

    V16 au, bu[NT];
    au.h[0] = *(const v8bf*)(Ap);
    au.h[1] = *(const v8bf*)(Ap + 16);
#pragma unroll
    for (int t = 0; t < NT; ++t) {
        for (int s = 0; s < 8; ++s) {
            bu[t].v[s]     = Wp[(size_t)s * N + t * 16];
            bu[t].v[s + 8] = Wp[(size_t)(s + 16) * N + t * 16];
        }
    }

    for (int k0 = 32; k0 < K; k0 += 32) {
        Ap += 32;
        Wp += wstep;
        __builtin_prefetch(Wp + wstep, 0, 1);   // -> global_prefetch_b8 (speculative)
        V16 an, bn[NT];
        an.h[0] = *(const v8bf*)(Ap);
        an.h[1] = *(const v8bf*)(Ap + 16);
#pragma unroll
        for (int t = 0; t < NT; ++t) {
            for (int s = 0; s < 8; ++s) {
                bn[t].v[s]     = Wp[(size_t)s * N + t * 16];
                bn[t].v[s + 8] = Wp[(size_t)(s + 16) * N + t * 16];
            }
        }
#pragma unroll
        for (int t = 0; t < NT; ++t)
            acc[t] = __builtin_amdgcn_wmma_f32_16x16x32_bf16(
                false, au.v, false, bu[t].v, (short)0, acc[t], false, false);
        au = an;
#pragma unroll
        for (int t = 0; t < NT; ++t) bu[t] = bn[t];
    }
#pragma unroll
    for (int t = 0; t < NT; ++t)
        acc[t] = __builtin_amdgcn_wmma_f32_16x16x32_bf16(
            false, au.v, false, bu[t].v, (short)0, acc[t], false, false);

    // C/D layout: lane<16: N=lane, M=v ; lane>=16: N=lane-16, M=8+v
#pragma unroll
    for (int t = 0; t < NT; ++t) {
#pragma unroll
        for (int v = 0; v < 8; ++v) {
            const int r = m0 + half * 8 + v;
            const int c = n0 + t * 16 + lm;
            float x = acc[t][v];
            if (bias)  x += bias[c];
            if (resid) x += resid[(size_t)r * N + c];
            if (relu)  x = fmaxf(x, 0.0f);
            if (Cf) Cf[(size_t)r * N + c] = x;
            if (Cb) Cb[(size_t)r * N + c] = (bf16_t)x;
        }
    }
}

// ======================= elementwise add + f32->bf16 =====================
__global__ __launch_bounds__(256)
void cvt_add_kernel(const float* __restrict__ a, const float* __restrict__ b,
                    bf16_t* __restrict__ o, int n)
{
    int i = blockIdx.x * 256 + threadIdx.x;
    if (i < n) {
        float v = a[i];
        if (b) v += b[i];
        o[i] = (bf16_t)v;
    }
}

// ======================= LayerNorm (row = 256) ===========================
__global__ __launch_bounds__(256)
void ln_kernel(const float* __restrict__ x, const float* __restrict__ g,
               const float* __restrict__ bt, float* __restrict__ outf,
               bf16_t* __restrict__ outb, int Mr)
{
    const int wave = threadIdx.x >> 5, lane = threadIdx.x & 31;
    const int row = blockIdx.x * 8 + wave;
    if (row >= Mr) return;
    const float* px = x + (size_t)row * Dm;
    float4 a = ((const float4*)px)[lane];
    float4 b = ((const float4*)px)[lane + 32];
    float s  = a.x + a.y + a.z + a.w + b.x + b.y + b.z + b.w;
    float sq = a.x*a.x + a.y*a.y + a.z*a.z + a.w*a.w
             + b.x*b.x + b.y*b.y + b.z*b.z + b.w*b.w;
#pragma unroll
    for (int o = 16; o; o >>= 1) {
        s  += __shfl_xor(s,  o, 32);
        sq += __shfl_xor(sq, o, 32);
    }
    const float mean = s * (1.0f / Dm);
    const float var  = sq * (1.0f / Dm) - mean * mean;
    const float inv  = rsqrtf(var + 1e-5f);
    const int c0 = lane * 4, c1 = 128 + lane * 4;
    float va[8] = { a.x, a.y, a.z, a.w, b.x, b.y, b.z, b.w };
#pragma unroll
    for (int j = 0; j < 8; ++j) {
        const int c = (j < 4) ? (c0 + j) : (c1 + j - 4);
        float y = (va[j] - mean) * inv * g[c] + bt[c];
        if (outf) outf[(size_t)row * Dm + c] = y;
        if (outb) outb[(size_t)row * Dm + c] = (bf16_t)y;
    }
}

// ======================= bilinear sample helper ==========================
// p points at val[b, level_base + 0, h, c]; pixel stride = NH*HD = 256 floats
__device__ __forceinline__ float bilin(const float* __restrict__ p,
                                       float x, float y, int H, int W)
{
    const float xf = floorf(x), yf = floorf(y);
    const float wx = x - xf,   wy = y - yf;
    const int   x0 = (int)xf,  y0 = (int)yf;
    float r = 0.0f;
#pragma unroll
    for (int dy = 0; dy < 2; ++dy) {
        const int yi = y0 + dy;
        if (yi < 0 || yi >= H) continue;
        const float wyv = dy ? wy : 1.0f - wy;
#pragma unroll
        for (int dx = 0; dx < 2; ++dx) {
            const int xi = x0 + dx;
            if (xi < 0 || xi >= W) continue;
            const float wxv = dx ? wx : 1.0f - wx;
            r += wyv * wxv * p[(size_t)(yi * W + xi) * (NH * HD)];
        }
    }
    return r;
}

// ================= self deformable attn sampling (1 level, 4 pts) ========
__global__ __launch_bounds__(256)
void msda_self_kernel(const float* __restrict__ val,     // [B, LQ, NH, HD]
                      const float* __restrict__ off,     // [Mq, 64]
                      const float* __restrict__ attnlog, // [Mq, 32]
                      const float* __restrict__ ref2,    // [B, LQ, 1, 2]
                      bf16_t* __restrict__ out)          // [Mq, 256]
{
    const int wave = threadIdx.x >> 5, lane = threadIdx.x & 31;
    const int g = blockIdx.x * 8 + wave;
    if (g >= Mq * NH) return;
    const int h   = g & (NH - 1);
    const int row = g >> 3;
    const int b   = row / LQ;

    const float* lgp = attnlog + (size_t)row * 32 + h * 4;
    float lg[4]; float mx = -1e30f;
#pragma unroll
    for (int i = 0; i < 4; ++i) { lg[i] = lgp[i]; mx = fmaxf(mx, lg[i]); }
    float s = 0.0f;
#pragma unroll
    for (int i = 0; i < 4; ++i) { lg[i] = __expf(lg[i] - mx); s += lg[i]; }
    const float inv = 1.0f / s;

    const float rx = ref2[(size_t)row * 2 + 0];
    const float ry = ref2[(size_t)row * 2 + 1];
    const float* offp = off + (size_t)row * 64 + h * 8;
    const float* vb = val + ((size_t)b * LQ * NH + h) * HD + lane;

    float acc = 0.0f;
#pragma unroll
    for (int p = 0; p < 4; ++p) {
        const float ox = offp[p * 2 + 0];
        const float oy = offp[p * 2 + 1];
        const float x = (rx + ox / (float)BEVW) * (float)BEVW - 0.5f;
        const float y = (ry + oy / (float)BEVH) * (float)BEVH - 0.5f;
        acc += lg[p] * inv * bilin(vb, x, y, BEVH, BEVW);
    }
    out[(size_t)row * Dm + h * HD + lane] = (bf16_t)acc;
}

// ================ cross deformable attn sampling (4 levels, 8 pts) =======
__global__ __launch_bounds__(256)
void msda_cross_kernel(const float* __restrict__ val,     // [B, LV, NH, HD]
                       const float* __restrict__ off,     // [Mq, 512]
                       const float* __restrict__ attnlog, // [Mq, 256]
                       const float* __restrict__ ref3,    // [B, LQ, 4, 2]
                       bf16_t* __restrict__ out)          // [Mq, 256]
{
    const int wave = threadIdx.x >> 5, lane = threadIdx.x & 31;
    const int g = blockIdx.x * 8 + wave;
    if (g >= Mq * NH) return;
    const int h   = g & (NH - 1);
    const int row = g >> 3;
    const int b   = row / LQ;

    const int Hs[4] = { 92, 46, 23, 12 };
    const int Ws[4] = { 160, 80, 40, 20 };
    const int St[4] = { 0, 14720, 18400, 19320 };

    const float* lgp = attnlog + (size_t)row * 256 + h * 32;
    float lg[32]; float mx = -1e30f;
#pragma unroll
    for (int i = 0; i < 32; ++i) { lg[i] = lgp[i]; mx = fmaxf(mx, lg[i]); }
    float s = 0.0f;
#pragma unroll
    for (int i = 0; i < 32; ++i) { lg[i] = __expf(lg[i] - mx); s += lg[i]; }
    const float inv = 1.0f / s;

    const float* offp = off + (size_t)row * 512 + h * 64;
    float acc = 0.0f;
#pragma unroll
    for (int l = 0; l < 4; ++l) {
        const int H = Hs[l], W = Ws[l];
        const float rx = ref3[(size_t)row * 8 + l * 2 + 0];
        const float ry = ref3[(size_t)row * 8 + l * 2 + 1];
        const float* vb = val + ((size_t)(b * LV + St[l]) * NH + h) * HD + lane;
#pragma unroll
        for (int p = 0; p < 8; ++p) {
            const float ox = offp[l * 16 + p * 2 + 0];
            const float oy = offp[l * 16 + p * 2 + 1];
            const float x = (rx + ox / (float)W) * (float)W - 0.5f;
            const float y = (ry + oy / (float)H) * (float)H - 0.5f;
            acc += lg[l * 8 + p] * inv * bilin(vb, x, y, H, W);
        }
    }
    out[(size_t)row * Dm + h * HD + lane] = (bf16_t)acc;
}

// ============================== host side ================================
extern "C" void kernel_launch(void* const* d_in, const int* in_sizes, int n_in,
                              void* d_out, int out_size, void* d_ws, size_t ws_size,
                              hipStream_t stream)
{
    const float* query   = (const float*)d_in[0];
    const float* value   = (const float*)d_in[1];
    const float* bev_pos = (const float*)d_in[2];
    const float* ref2d   = (const float*)d_in[3];
    const float* ref3d   = (const float*)d_in[4];
    const float* Woff_s  = (const float*)d_in[5];
    const float* boff_s  = (const float*)d_in[6];
    const float* Wattn_s = (const float*)d_in[7];
    const float* battn_s = (const float*)d_in[8];
    const float* Wv_s    = (const float*)d_in[9];
    const float* bv_s    = (const float*)d_in[10];
    const float* Wo_s    = (const float*)d_in[11];
    const float* bo_s    = (const float*)d_in[12];
    const float* Woff_c  = (const float*)d_in[13];
    const float* boff_c  = (const float*)d_in[14];
    const float* Wattn_c = (const float*)d_in[15];
    const float* battn_c = (const float*)d_in[16];
    const float* Wv_c    = (const float*)d_in[17];
    const float* bv_c    = (const float*)d_in[18];
    const float* Wo_c    = (const float*)d_in[19];
    const float* bo_c    = (const float*)d_in[20];
    const float* W1      = (const float*)d_in[21];
    const float* b1f     = (const float*)d_in[22];
    const float* W2      = (const float*)d_in[23];
    const float* b2f     = (const float*)d_in[24];
    const float* ln1_g   = (const float*)d_in[25];
    const float* ln1_b   = (const float*)d_in[26];
    const float* ln2_g   = (const float*)d_in[27];
    const float* ln2_b   = (const float*)d_in[28];
    const float* ln3_g   = (const float*)d_in[29];
    const float* ln3_b   = (const float*)d_in[30];
    // d_in[31..34] (spatial_shapes/level_start/bev_h/bev_w) are fixed constants.

    char*  ws = (char*)d_ws;
    size_t o  = 0;
    auto alloc = [&](size_t bytes) -> char* {
        char* p = ws + o;
        o = (o + bytes + 255) & ~(size_t)255;
        return p;
    };

    auto cvt = [&](const float* src, bf16_t* dst, long n) {
        cvt_add_kernel<<<dim3((unsigned)((n + 255) / 256)), dim3(256), 0, stream>>>(
            src, nullptr, dst, (int)n);
    };
    auto gemm = [&](const bf16_t* A, const bf16_t* W, const float* bias,
                    const float* resid, float* Cf, bf16_t* Cb,
                    int Mr, int K, int N, int relu) {
        const unsigned gx = (unsigned)((Mr / 16 + 7) / 8);
        if (N % 64 == 0) {
            dim3 grid(gx, (unsigned)(N / 64));
            gemm_bf16_kernel<4><<<grid, 256, 0, stream>>>(A, W, bias, resid, Cf, Cb, Mr, K, N, relu);
        } else {
            dim3 grid(gx, (unsigned)(N / 32));
            gemm_bf16_kernel<2><<<grid, 256, 0, stream>>>(A, W, bias, resid, Cf, Cb, Mr, K, N, relu);
        }
    };
    auto ln = [&](const float* x, const float* g, const float* b,
                  float* outf, bf16_t* outb) {
        ln_kernel<<<dim3((unsigned)((Mq + 7) / 8)), dim3(256), 0, stream>>>(x, g, b, outf, outb, Mq);
    };

    // ---------------- persistent allocations ----------------
    bf16_t* wWoff_s  = (bf16_t*)alloc((size_t)Dm * 64  * 2);
    bf16_t* wWattn_s = (bf16_t*)alloc((size_t)Dm * 32  * 2);
    bf16_t* wWv_s    = (bf16_t*)alloc((size_t)Dm * Dm  * 2);
    bf16_t* wWo_s    = (bf16_t*)alloc((size_t)Dm * Dm  * 2);
    bf16_t* wWoff_c  = (bf16_t*)alloc((size_t)Dm * 512 * 2);
    bf16_t* wWattn_c = (bf16_t*)alloc((size_t)Dm * 256 * 2);
    bf16_t* wWv_c    = (bf16_t*)alloc((size_t)Dm * Dm  * 2);
    bf16_t* wWo_c    = (bf16_t*)alloc((size_t)Dm * Dm  * 2);
    bf16_t* wW1      = (bf16_t*)alloc((size_t)Dm * DFF * 2);
    bf16_t* wW2      = (bf16_t*)alloc((size_t)DFF * Dm * 2);
    float*  x1       = (float*) alloc((size_t)Mq * Dm * 4);
    float*  x2       = (float*) alloc((size_t)Mq * Dm * 4);
    float*  x3       = (float*) alloc((size_t)Mq * Dm * 4);
    const size_t poolBase = o;

    // ---------------- weight conversion (tiny) ----------------
    cvt(Woff_s,  wWoff_s,  (long)Dm * 64);
    cvt(Wattn_s, wWattn_s, (long)Dm * 32);
    cvt(Wv_s,    wWv_s,    (long)Dm * Dm);
    cvt(Wo_s,    wWo_s,    (long)Dm * Dm);
    cvt(Woff_c,  wWoff_c,  (long)Dm * 512);
    cvt(Wattn_c, wWattn_c, (long)Dm * 256);
    cvt(Wv_c,    wWv_c,    (long)Dm * Dm);
    cvt(Wo_c,    wWo_c,    (long)Dm * Dm);
    cvt(W1,      wW1,      (long)Dm * DFF);
    cvt(W2,      wW2,      (long)DFF * Dm);

    const unsigned sampBlocks = (unsigned)((Mq * NH + 7) / 8);

    // ================= stage 1: self deformable attn =================
    o = poolBase;
    bf16_t* qpos_b  = (bf16_t*)alloc((size_t)Mq * Dm * 2);
    bf16_t* query_b = (bf16_t*)alloc((size_t)Mq * Dm * 2);
    float*  val_s   = (float*) alloc((size_t)Mq * Dm * 4);
    float*  off_s   = (float*) alloc((size_t)Mq * 64 * 4);
    float*  attn_s  = (float*) alloc((size_t)Mq * 32 * 4);
    bf16_t* samp_s  = (bf16_t*)alloc((size_t)Mq * Dm * 2);

    cvt_add_kernel<<<dim3((unsigned)(((long)Mq * Dm + 255) / 256)), dim3(256), 0, stream>>>(
        query, bev_pos, qpos_b, Mq * Dm);
    cvt(query, query_b, (long)Mq * Dm);
    gemm(query_b, wWv_s,    bv_s,    nullptr, val_s,  nullptr, Mq, Dm, Dm, 0);
    gemm(qpos_b,  wWoff_s,  boff_s,  nullptr, off_s,  nullptr, Mq, Dm, 64, 0);
    gemm(qpos_b,  wWattn_s, battn_s, nullptr, attn_s, nullptr, Mq, Dm, 32, 0);
    msda_self_kernel<<<dim3(sampBlocks), dim3(256), 0, stream>>>(val_s, off_s, attn_s, ref2d, samp_s);
    gemm(samp_s, wWo_s, bo_s, query, x1, nullptr, Mq, Dm, Dm, 0);   // + residual(query)

    // ================= stage 2: cross deformable attn ================
    o = poolBase;
    bf16_t* x1n_b   = (bf16_t*)alloc((size_t)Mq * Dm * 2);
    bf16_t* value_b = (bf16_t*)alloc((size_t)Mv * Dm * 2);
    float*  val_c   = (float*) alloc((size_t)Mv * Dm * 4);
    float*  off_c   = (float*) alloc((size_t)Mq * 512 * 4);
    float*  attn_c  = (float*) alloc((size_t)Mq * 256 * 4);
    bf16_t* samp_c  = (bf16_t*)alloc((size_t)Mq * Dm * 2);

    ln(x1, ln1_g, ln1_b, nullptr, x1n_b);
    cvt(value, value_b, (long)Mv * Dm);
    gemm(value_b, wWv_c,    bv_c,    nullptr, val_c,  nullptr, Mv, Dm, Dm,  0);
    gemm(x1n_b,   wWoff_c,  boff_c,  nullptr, off_c,  nullptr, Mq, Dm, 512, 0);
    gemm(x1n_b,   wWattn_c, battn_c, nullptr, attn_c, nullptr, Mq, Dm, 256, 0);
    msda_cross_kernel<<<dim3(sampBlocks), dim3(256), 0, stream>>>(val_c, off_c, attn_c, ref3d, samp_c);
    gemm(samp_c, wWo_c, bo_c, x1, x2, nullptr, Mq, Dm, Dm, 0);      // + residual(x1)

    // ================= stage 3: FFN + final LN =======================
    o = poolBase;
    bf16_t* x3_b = (bf16_t*)alloc((size_t)Mq * Dm * 2);
    bf16_t* h_b  = (bf16_t*)alloc((size_t)Mq * DFF * 2);
    float*  x4   = (float*) alloc((size_t)Mq * Dm * 4);

    ln(x2, ln2_g, ln2_b, x3, x3_b);
    gemm(x3_b, wW1, b1f, nullptr, nullptr, h_b, Mq, Dm, DFF, 1);    // ReLU, bf16 out
    gemm(h_b,  wW2, b2f, x3, x4, nullptr, Mq, DFF, Dm, 0);          // + residual(x3)
    ln(x4, ln3_g, ln3_b, (float*)d_out, nullptr);
}
